// SpatialGraphTransformer_26895085208198
// MI455X (gfx1250) — matrix operand order repaired and checked by reading, not compiled, not run on hardware
//
#include <hip/hip_runtime.h>
#include <hip/hip_fp16.h>

typedef __attribute__((ext_vector_type(16))) _Float16 v16h;
typedef __attribute__((ext_vector_type(8)))  _Float16 v8h;
typedef __attribute__((ext_vector_type(8)))  float    v8f;

union V16H { v16h v; v8h h[2]; };

// ---------------------------------------------------------------------------
// Generic NT GEMM on f16 operands, f32 accumulate via v_wmma_f32_16x16x32_f16.
//   C[m][n] = alpha * sum_k A[m][k] * Bt[n][k]  (+ residual) (then optional silu)
// Each wave owns a 32x64 C strip (2 M-tiles x 4 N-tiles = 8 WMMAs per K-step,
// B fragments reused across both M-tiles). Register double-buffering: next
// K-step's fragments are issued before the current K-step's WMMAs so VMEM
// latency overlaps matrix math instead of s_wait_loadcnt 0 stalls.
// Batched over blockIdx.z = b*Z2 + h; optional device-side expert selection:
// Bt += cur[b]*curStride.
// Requires: M % 256 == 0 (grid.x = M/256), K % 32 == 0, 16B/32B alignment.
// ---------------------------------------------------------------------------
__global__ __launch_bounds__(256) void gemm_wmma_nt(
    const _Float16* __restrict__ A,  long sAb, long sAh, int lda,
    const _Float16* __restrict__ Bt, long sBb, long sBh, int ldb,
    float* __restrict__ C32, _Float16* __restrict__ C16, long sCb, long sCh, int ldc,
    const float* __restrict__ Res,
    int M, int N, int K, float alpha, int act,
    const int* __restrict__ cur, long curStride, int Z2)
{
    int z  = blockIdx.z;
    int bb = z / Z2;
    int hh = z - bb * Z2;

    const _Float16* Ab = A + (size_t)bb * sAb + (size_t)hh * sAh;
    long boff = (long)bb * sBb + (long)hh * sBh;
    if (cur) boff += (long)cur[bb] * curStride;
    const _Float16* Bb = Bt + boff;
    long coff = (long)bb * sCb + (long)hh * sCh;

    int lane = threadIdx.x & 31;
    int wv   = threadIdx.x >> 5;              // 0..7
    int m0   = blockIdx.x * 256 + wv * 32;    // two 16-row tiles: m0, m0+16
    int n0   = blockIdx.y * 64;
    if (m0 >= M) return;

    // A fragment addressing (contiguous k order k = 16h + s, matching B):
    //   lanes 0-15 : row m, k in [0,16)
    //   lanes 16-31: row m, k in [16,32)
    int aks = (lane >> 4) * 16;
    // B fragment addressing (contiguous k order k = 16h + s, matching A):
    //   lanes 0-15 : col n, k in [0,16) ; lanes 16-31: col n, k in [16,32)
    int bks = (lane >> 4) * 16;

    const _Float16* arow0 = Ab + (size_t)(m0 + (lane & 15)) * lda;
    const _Float16* arow1 = arow0 + (size_t)16 * lda;
    const _Float16* b0    = Bb + (size_t)(n0 + (lane & 15)) * ldb + bks;
    const _Float16* b1    = b0 + (size_t)16 * ldb;
    const _Float16* b2    = b0 + (size_t)32 * ldb;
    const _Float16* b3    = b0 + (size_t)48 * ldb;

    v8f acc00 = {}, acc01 = {}, acc02 = {}, acc03 = {};
    v8f acc10 = {}, acc11 = {}, acc12 = {}, acc13 = {};

    // ---- preload K-step 0 fragments
    V16H a0, a1;
    a0.h[0] = *(const v8h*)(arow0 + aks);
    a0.h[1] = *(const v8h*)(arow0 + aks + 8);
    a1.h[0] = *(const v8h*)(arow1 + aks);
    a1.h[1] = *(const v8h*)(arow1 + aks + 8);
    v16h fb0 = *(const v16h*)(b0);
    v16h fb1 = *(const v16h*)(b1);
    v16h fb2 = *(const v16h*)(b2);
    v16h fb3 = *(const v16h*)(b3);

    // ---- main loop: issue next loads, then WMMAs on current fragments
    for (int kb = 0; kb + 32 < K; kb += 32) {
        int kn = kb + 32;
        if (kn + 32 < K) {                        // keep L2 ahead
            __builtin_prefetch(arow0 + kn + 32 + aks, 0, 3);
            __builtin_prefetch(b0 + kn + 32, 0, 3);
            __builtin_prefetch(b2 + kn + 32, 0, 3);
        }
        V16H na0, na1;
        na0.h[0] = *(const v8h*)(arow0 + kn + aks);
        na0.h[1] = *(const v8h*)(arow0 + kn + aks + 8);
        na1.h[0] = *(const v8h*)(arow1 + kn + aks);
        na1.h[1] = *(const v8h*)(arow1 + kn + aks + 8);
        v16h nb0 = *(const v16h*)(b0 + kn);
        v16h nb1 = *(const v16h*)(b1 + kn);
        v16h nb2 = *(const v16h*)(b2 + kn);
        v16h nb3 = *(const v16h*)(b3 + kn);

        acc00 = __builtin_amdgcn_wmma_f32_16x16x32_f16(false, a0.v, false, fb0, (short)0, acc00, false, false);
        acc10 = __builtin_amdgcn_wmma_f32_16x16x32_f16(false, a1.v, false, fb0, (short)0, acc10, false, false);
        acc01 = __builtin_amdgcn_wmma_f32_16x16x32_f16(false, a0.v, false, fb1, (short)0, acc01, false, false);
        acc11 = __builtin_amdgcn_wmma_f32_16x16x32_f16(false, a1.v, false, fb1, (short)0, acc11, false, false);
        acc02 = __builtin_amdgcn_wmma_f32_16x16x32_f16(false, a0.v, false, fb2, (short)0, acc02, false, false);
        acc12 = __builtin_amdgcn_wmma_f32_16x16x32_f16(false, a1.v, false, fb2, (short)0, acc12, false, false);
        acc03 = __builtin_amdgcn_wmma_f32_16x16x32_f16(false, a0.v, false, fb3, (short)0, acc03, false, false);
        acc13 = __builtin_amdgcn_wmma_f32_16x16x32_f16(false, a1.v, false, fb3, (short)0, acc13, false, false);

        a0 = na0; a1 = na1; fb0 = nb0; fb1 = nb1; fb2 = nb2; fb3 = nb3;
    }
    // ---- final K-step
    acc00 = __builtin_amdgcn_wmma_f32_16x16x32_f16(false, a0.v, false, fb0, (short)0, acc00, false, false);
    acc10 = __builtin_amdgcn_wmma_f32_16x16x32_f16(false, a1.v, false, fb0, (short)0, acc10, false, false);
    acc01 = __builtin_amdgcn_wmma_f32_16x16x32_f16(false, a0.v, false, fb1, (short)0, acc01, false, false);
    acc11 = __builtin_amdgcn_wmma_f32_16x16x32_f16(false, a1.v, false, fb1, (short)0, acc11, false, false);
    acc02 = __builtin_amdgcn_wmma_f32_16x16x32_f16(false, a0.v, false, fb2, (short)0, acc02, false, false);
    acc12 = __builtin_amdgcn_wmma_f32_16x16x32_f16(false, a1.v, false, fb2, (short)0, acc12, false, false);
    acc03 = __builtin_amdgcn_wmma_f32_16x16x32_f16(false, a0.v, false, fb3, (short)0, acc03, false, false);
    acc13 = __builtin_amdgcn_wmma_f32_16x16x32_f16(false, a1.v, false, fb3, (short)0, acc13, false, false);

    // C/D layout: VGPR r, lanes 0-15 -> (mt+r, n0+lane); lanes 16-31 -> (mt+8+r, n0+lane-16)
    auto epilogue = [&](v8f acc, int mt, int t) {
        int ncol = n0 + t * 16 + (lane & 15);
        if (ncol >= N) return;
        int mbase = m0 + mt * 16 + ((lane >> 4) << 3);
        #pragma unroll
        for (int r = 0; r < 8; ++r) {
            size_t cidx = (size_t)coff + (size_t)(mbase + r) * ldc + ncol;
            float v = acc[r] * alpha;
            if (Res) v += Res[cidx];
            if (act == 1) v = v * (1.0f / (1.0f + __expf(-v)));   // silu
            if (C32) C32[cidx] = v;
            if (C16) C16[cidx] = (_Float16)v;
        }
    };
    epilogue(acc00, 0, 0); epilogue(acc01, 0, 1); epilogue(acc02, 0, 2); epilogue(acc03, 0, 3);
    epilogue(acc10, 1, 0); epilogue(acc11, 1, 1); epilogue(acc12, 1, 2); epilogue(acc13, 1, 3);
}

// ---------------------------------------------------------------------------
// Transpose + convert: in (K,N) f32  ->  out (N,K) f16, batched over z.
// ---------------------------------------------------------------------------
__global__ __launch_bounds__(256) void transpose_f32_to_f16(
    const float* __restrict__ in, _Float16* __restrict__ out,
    int K, int N, long inStride, long outStride)
{
    __shared__ float tile[32][33];
    const float* ib = in  + (size_t)blockIdx.z * inStride;
    _Float16*    ob = out + (size_t)blockIdx.z * outStride;
    int k0 = blockIdx.y * 32, n0 = blockIdx.x * 32;
    #pragma unroll
    for (int i = 0; i < 4; ++i) {
        int k = k0 + threadIdx.y + 8 * i;
        int n = n0 + threadIdx.x;
        if (k < K && n < N) tile[threadIdx.y + 8 * i][threadIdx.x] = ib[(size_t)k * N + n];
    }
    __syncthreads();
    #pragma unroll
    for (int i = 0; i < 4; ++i) {
        int n = n0 + threadIdx.y + 8 * i;
        int k = k0 + threadIdx.x;
        if (k < K && n < N) ob[(size_t)n * K + k] = (_Float16)tile[threadIdx.x][threadIdx.y + 8 * i];
    }
}

// ---------------------------------------------------------------------------
__global__ void embed_kernel(const int* __restrict__ ids, const float* __restrict__ emb,
                             float* __restrict__ x, int D, float scale)
{
    int row = blockIdx.x;
    int id  = ids[row];
    for (int d = threadIdx.x; d < D; d += blockDim.x)
        x[(size_t)row * D + d] = emb[(size_t)id * D + d] * scale;
}

// RMSNorm row -> f16 output, optional per-batch expert-selected weight.
__global__ __launch_bounds__(256) void rmsnorm_h(
    const float* __restrict__ x, const float* __restrict__ w,
    const int* __restrict__ cur, long wSel, int rowsPerBatch,
    _Float16* __restrict__ out, int D, float eps)
{
    __shared__ float red[256];
    int row = blockIdx.x;
    const float* xr = x + (size_t)row * D;
    const float* wr = w;
    if (cur) wr += (size_t)cur[row / rowsPerBatch] * wSel;
    float ss = 0.f;
    for (int d = threadIdx.x; d < D; d += blockDim.x) { float v = xr[d]; ss += v * v; }
    red[threadIdx.x] = ss; __syncthreads();
    for (int off = 128; off > 0; off >>= 1) {
        if ((int)threadIdx.x < off) red[threadIdx.x] += red[threadIdx.x + off];
        __syncthreads();
    }
    float r = rsqrtf(red[0] / D + eps);
    for (int d = threadIdx.x; d < D; d += blockDim.x)
        out[(size_t)row * D + d] = (_Float16)(xr[d] * r * wr[d]);
}

// RoPE on q,k (f32 in, f16 out), layout (B,S,D) with heads contiguous (hd=64).
__global__ void rope_h(const float* __restrict__ q, const float* __restrict__ k,
                       _Float16* __restrict__ qh, _Float16* __restrict__ kh,
                       int S, int D, int hd, float theta, int total)
{
    int idx = blockIdx.x * blockDim.x + threadIdx.x;
    if (idx >= total) return;
    int half = D / 2;
    int p    = idx % half;
    int row  = idx / half;            // b*S + s
    int s    = row % S;
    int h    = p / (hd / 2);
    int i    = p % (hd / 2);
    float ang = (float)s * __powf(theta, -(2.0f * i) / (float)hd);
    float c = __cosf(ang), sn = __sinf(ang);
    size_t base = (size_t)row * D + h * hd + 2 * i;
    float q1 = q[base], q2 = q[base + 1];
    qh[base]     = (_Float16)(q1 * c - q2 * sn);
    qh[base + 1] = (_Float16)(q1 * sn + q2 * c);
    float k1 = k[base], k2 = k[base + 1];
    kh[base]     = (_Float16)(k1 * c - k2 * sn);
    kh[base + 1] = (_Float16)(k1 * sn + k2 * c);
}

// v (B,S,D) f32 -> vT (B,H,hd,S) f16   (so the O-GEMM is NT with K=S contiguous)
__global__ void vtrans_h(const float* __restrict__ v, _Float16* __restrict__ vT,
                         int H, int S, int hd, int total)
{
    int idx = blockIdx.x * blockDim.x + threadIdx.x;
    if (idx >= total) return;
    int s = idx % S;
    int d = (idx / S) % hd;
    int h = (idx / (S * hd)) % H;
    int b = idx / (S * hd * H);
    vT[idx] = (_Float16)v[((size_t)(b * S + s)) * (H * hd) + h * hd + d];
}

// Causal softmax over rows of (B,H,S,S) scores (scale already applied) -> f16 att.
__global__ __launch_bounds__(256) void softmax_causal_h(
    const float* __restrict__ scores, _Float16* __restrict__ att, int S)
{
    __shared__ float red[256];
    int z  = blockIdx.x;              // (b*H+h)*S + qi
    int qi = z % S;
    int j  = threadIdx.x;
    const float* row = scores + (size_t)z * S;
    float v = (j <= qi) ? row[j] : -1e30f;
    red[j] = v; __syncthreads();
    for (int off = 128; off > 0; off >>= 1) {
        if (j < off) red[j] = fmaxf(red[j], red[j + off]);
        __syncthreads();
    }
    float mx = red[0]; __syncthreads();
    float e = (j <= qi) ? __expf(v - mx) : 0.f;
    red[j] = e; __syncthreads();
    for (int off = 128; off > 0; off >>= 1) {
        if (j < off) red[j] += red[j + off];
        __syncthreads();
    }
    att[(size_t)z * S + j] = (_Float16)(e / red[0]);
}

// Mean-pool + gate softmax + aux accumulation + adjacency-masked argmax routing.
__global__ __launch_bounds__(256) void routing_kernel(
    const float* __restrict__ x, const float* __restrict__ gate_w,
    float* __restrict__ accProbs, float* __restrict__ accLb, int* __restrict__ cur,
    int B, int S, int D, int C)
{
    __shared__ float pooled[2][512];
    __shared__ float sc[2][8];
    __shared__ float pr[2][8];
    int t = threadIdx.x;
    for (int b = 0; b < B; ++b)
        for (int d = t; d < D; d += 256) {
            float s = 0.f;
            for (int si = 0; si < S; ++si) s += x[((size_t)b * S + si) * D + d];
            pooled[b][d] = s / (float)S;
        }
    __syncthreads();
    int NC = C + 1;
    if (t < B * NC) {
        int b = t / NC, j = t % NC;
        float s = 0.f;
        for (int d = 0; d < D; ++d) s += pooled[b][d] * gate_w[d * NC + j];
        sc[b][j] = s;
    }
    __syncthreads();
    if (t < B) {
        float mx = sc[t][0];
        for (int j = 1; j < NC; ++j) mx = fmaxf(mx, sc[t][j]);
        float sum = 0.f;
        for (int j = 0; j < NC; ++j) { float e = __expf(sc[t][j] - mx); pr[t][j] = e; sum += e; }
        for (int j = 0; j < NC; ++j) pr[t][j] /= sum;
    }
    __syncthreads();
    if (t == 0) {
        float lb = 0.f;
        for (int j = 0; j < NC; ++j) {
            float avg = 0.f;
            for (int b = 0; b < B; ++b) { accProbs[j] += pr[b][j]; avg += pr[b][j]; }
            avg /= (float)B;
            lb += avg * avg;
        }
        accLb[0] += (float)(C + 1) * lb;
        for (int b = 0; b < B; ++b) {     // argmax over ring-adjacency of OLD cur
            int c = cur[b];
            int a1 = (c + 1) % C, a2 = (c + 2) % C;
            int best = 0; float bv = -1e38f;
            for (int j = 0; j < C; ++j) {
                float v = (j == a1 || j == a2) ? sc[b][j] : -1e30f;
                if (v > bv) { bv = v; best = j; }
            }
            cur[b] = best;
        }
    }
}

__global__ void init_kernel(const int* __restrict__ start, int* __restrict__ cur,
                            float* __restrict__ accProbs, float* __restrict__ accLb,
                            int B, int NC)
{
    int t = threadIdx.x;
    if (t < B)  cur[t] = start[t];
    if (t < NC) accProbs[t] = 0.f;
    if (t == 0) accLb[0] = 0.f;
}

__global__ void aux_kernel(const float* __restrict__ accProbs, const float* __restrict__ accLb,
                           float* __restrict__ out, int C, int B, int steps)
{
    if (threadIdx.x == 0 && blockIdx.x == 0) {
        int NC = C + 1;
        float denom = (float)(steps * B);
        float mean = 0.f, m2 = 0.f;
        for (int j = 0; j < NC; ++j) { float p = accProbs[j] / denom; mean += p; m2 += p * p; }
        mean /= NC; m2 /= NC;
        out[0] = m2 - mean * mean;        // population variance
        out[1] = accLb[0] / (float)steps;
    }
}

// ---------------------------------------------------------------------------
static inline void launch_gemm(hipStream_t st,
    const _Float16* A, long sAb, long sAh, int lda,
    const _Float16* Bt, long sBb, long sBh, int ldb,
    float* C32, _Float16* C16, long sCb, long sCh, int ldc,
    const float* Res, int M, int N, int K, float alpha, int act,
    const int* cur, long curStride, int Z, int Z2)
{
    dim3 g(M / 256, (N + 63) / 64, Z), b(256);
    gemm_wmma_nt<<<g, b, 0, st>>>(A, sAb, sAh, lda, Bt, sBb, sBh, ldb,
                                  C32, C16, sCb, sCh, ldc, Res,
                                  M, N, K, alpha, act, cur, curStride, Z2);
}

extern "C" void kernel_launch(void* const* d_in, const int* in_sizes, int n_in,
                              void* d_out, int out_size, void* d_ws, size_t ws_size,
                              hipStream_t stream)
{
    (void)in_sizes; (void)n_in; (void)out_size; (void)ws_size;
    const int B = 2, S = 256, D = 512, F = 2048, C = 4, H = 8, HD = 64, V = 32000;
    const int NC = C + 1, STEPS = 2;

    const int*   ids    = (const int*)  d_in[0];
    const int*   start  = (const int*)  d_in[1];
    const float* emb    = (const float*)d_in[2];
    const float* Wq     = (const float*)d_in[3];
    const float* Wk     = (const float*)d_in[4];
    const float* Wv     = (const float*)d_in[5];
    const float* Wo     = (const float*)d_in[6];
    const float* W1     = (const float*)d_in[7];
    const float* W2     = (const float*)d_in[8];
    const float* n1     = (const float*)d_in[9];
    const float* n2     = (const float*)d_in[10];
    const float* gatew  = (const float*)d_in[11];
    const float* fnorm  = (const float*)d_in[12];
    const float* fcw    = (const float*)d_in[13];
    float* logits = (float*)d_out;
    float* auxOut = logits + (size_t)B * S * V;

    // workspace allocator (256B aligned)
    char* p = (char*)d_ws;
    auto alloc = [&](size_t bytes) { void* r = p; p += (bytes + 255) & ~(size_t)255; return r; };
    _Float16* WqT = (_Float16*)alloc((size_t)C * D * D * 2);
    _Float16* WkT = (_Float16*)alloc((size_t)C * D * D * 2);
    _Float16* WvT = (_Float16*)alloc((size_t)C * D * D * 2);
    _Float16* WoT = (_Float16*)alloc((size_t)C * D * D * 2);
    _Float16* W1T = (_Float16*)alloc((size_t)C * F * D * 2);
    _Float16* W2T = (_Float16*)alloc((size_t)C * D * F * 2);
    _Float16* fcT = (_Float16*)alloc((size_t)V * D * 2);
    _Float16* xh  = (_Float16*)alloc((size_t)B * S * D * 2);
    _Float16* qh  = (_Float16*)alloc((size_t)B * S * D * 2);
    _Float16* kh  = (_Float16*)alloc((size_t)B * S * D * 2);
    _Float16* vTh = (_Float16*)alloc((size_t)B * H * HD * S * 2);
    _Float16* atth= (_Float16*)alloc((size_t)B * H * S * S * 2);
    _Float16* oh  = (_Float16*)alloc((size_t)B * S * D * 2);
    _Float16* hh  = (_Float16*)alloc((size_t)B * S * D * 2);
    _Float16* ffh = (_Float16*)alloc((size_t)B * S * F * 2);
    _Float16* fh  = (_Float16*)alloc((size_t)B * S * D * 2);
    float* x   = (float*)alloc((size_t)B * S * D * 4);
    float* q32 = (float*)alloc((size_t)B * S * D * 4);
    float* k32 = (float*)alloc((size_t)B * S * D * 4);
    float* v32 = (float*)alloc((size_t)B * S * D * 4);
    float* scb = (float*)alloc((size_t)B * H * S * S * 4);
    float* accP = (float*)alloc(8 * 4);
    float* accL = (float*)alloc(4);
    int*   cur  = (int*)alloc(8 * 4);

    dim3 tb(32, 8);
    // weight conversion + transpose (f32 (K,N) -> f16 (N,K))
    transpose_f32_to_f16<<<dim3(D/32, D/32, C), tb, 0, stream>>>(Wq, WqT, D, D, (long)D*D, (long)D*D);
    transpose_f32_to_f16<<<dim3(D/32, D/32, C), tb, 0, stream>>>(Wk, WkT, D, D, (long)D*D, (long)D*D);
    transpose_f32_to_f16<<<dim3(D/32, D/32, C), tb, 0, stream>>>(Wv, WvT, D, D, (long)D*D, (long)D*D);
    transpose_f32_to_f16<<<dim3(D/32, D/32, C), tb, 0, stream>>>(Wo, WoT, D, D, (long)D*D, (long)D*D);
    transpose_f32_to_f16<<<dim3(F/32, D/32, C), tb, 0, stream>>>(W1, W1T, D, F, (long)D*F, (long)D*F);
    transpose_f32_to_f16<<<dim3(D/32, F/32, C), tb, 0, stream>>>(W2, W2T, F, D, (long)D*F, (long)D*F);
    transpose_f32_to_f16<<<dim3(V/32, D/32, 1), tb, 0, stream>>>(fcw, fcT, D, V, 0, 0);

    init_kernel<<<1, 64, 0, stream>>>(start, cur, accP, accL, B, NC);
    embed_kernel<<<B * S, 256, 0, stream>>>(ids, emb, x, D, sqrtf((float)D));

    const float isq = 1.0f / sqrtf((float)HD);
    for (int step = 0; step < STEPS; ++step) {
        // xn = rmsnorm(x, n1[cur])  -> f16
        rmsnorm_h<<<B * S, 256, 0, stream>>>(x, n1, cur, (long)D, S, xh, D, 1e-6f);
        // q,k,v = xn @ W{q,k,v}[cur]   (M=256,N=512,K=512; batch=B via cur)
        launch_gemm(stream, xh, (long)S*D, 0, D, WqT, 0, 0, D, q32, nullptr, (long)S*D, 0, D,
                    nullptr, S, D, D, 1.f, 0, cur, (long)D*D, B, 1);
        launch_gemm(stream, xh, (long)S*D, 0, D, WkT, 0, 0, D, k32, nullptr, (long)S*D, 0, D,
                    nullptr, S, D, D, 1.f, 0, cur, (long)D*D, B, 1);
        launch_gemm(stream, xh, (long)S*D, 0, D, WvT, 0, 0, D, v32, nullptr, (long)S*D, 0, D,
                    nullptr, S, D, D, 1.f, 0, cur, (long)D*D, B, 1);
        // RoPE(q,k) -> f16 ; v -> (B,H,hd,S) f16
        rope_h<<<(B*S*D/2 + 255)/256, 256, 0, stream>>>(q32, k32, qh, kh, S, D, HD, 10000.f, B*S*D/2);
        vtrans_h<<<(B*H*HD*S + 255)/256, 256, 0, stream>>>(v32, vTh, H, S, HD, B*H*HD*S);
        // scores = q kᵀ / sqrt(hd)   (batch (b,h); M=N=256, K=64)
        launch_gemm(stream, qh, (long)S*D, (long)HD, D, kh, (long)S*D, (long)HD, D,
                    scb, nullptr, (long)H*S*S, (long)S*S, S, nullptr,
                    S, S, HD, isq, 0, nullptr, 0, B*H, H);
        softmax_causal_h<<<B * H * S, S, 0, stream>>>(scb, atth, S);
        // o = att @ v   (M=256,N=64,K=256) -> f16 directly into (B,S,D)
        launch_gemm(stream, atth, (long)H*S*S, (long)S*S, S, vTh, (long)H*HD*S, (long)HD*S, S,
                    nullptr, oh, (long)S*D, (long)HD, D, nullptr,
                    S, HD, S, 1.f, 0, nullptr, 0, B*H, H);
        // x = x + o @ Wo[cur]   (in-place residual)
        launch_gemm(stream, oh, (long)S*D, 0, D, WoT, 0, 0, D, x, nullptr, (long)S*D, 0, D,
                    x, S, D, D, 1.f, 0, cur, (long)D*D, B, 1);
        // hn = rmsnorm(x, n2[cur]) -> f16
        rmsnorm_h<<<B * S, 256, 0, stream>>>(x, n2, cur, (long)D, S, hh, D, 1e-6f);
        // ff = silu(hn @ W1[cur]) -> f16   (M=256,N=2048,K=512)
        launch_gemm(stream, hh, (long)S*D, 0, D, W1T, 0, 0, D, nullptr, ffh, (long)S*F, 0, F,
                    nullptr, S, F, D, 1.f, 1, cur, (long)F*D, B, 1);
        // x = x + ff @ W2[cur]   (M=256,N=512,K=2048)
        launch_gemm(stream, ffh, (long)S*F, 0, F, W2T, 0, 0, F, x, nullptr, (long)S*D, 0, D,
                    x, S, D, F, 1.f, 0, cur, (long)D*F, B, 1);
        // gate + aux accumulation + routing update of cur
        routing_kernel<<<1, 256, 0, stream>>>(x, gatew, accP, accL, cur, B, S, D, C);
    }

    // logits = rmsnorm(x, final_norm) @ fc_out   (M=512,N=32000,K=512)
    rmsnorm_h<<<B * S, 256, 0, stream>>>(x, fnorm, nullptr, 0, S, fh, D, 1e-6f);
    launch_gemm(stream, fh, 0, 0, D, fcT, 0, 0, D, logits, nullptr, 0, 0, V,
                nullptr, B * S, V, D, 1.f, 0, nullptr, 0, 1, 1);
    aux_kernel<<<1, 32, 0, stream>>>(accP, accL, auxOut, C, B, STEPS);
}
